// GENConvBig_54932631716216
// MI455X (gfx1250) — matrix operand order repaired and checked by reading, not compiled, not run on hardware
//
#include <hip/hip_runtime.h>

#define NNODES 50000
#define NEDGES 800000
#define CCH 128
#define HCH 256
#define NC (NNODES * CCH)
#define BN_EPS 1e-5f
#define MSG_EPS 1e-7f
#define BETA 0.01f

typedef __attribute__((ext_vector_type(16))) __bf16 v16bf;
typedef __attribute__((ext_vector_type(8)))  float  v8f;

union BF16x16 { unsigned u[8]; v16bf v; };

__device__ inline unsigned short f2bf(float f) {
    unsigned u = __float_as_uint(f);
    u = (u + 0x7FFFu + ((u >> 16) & 1u)) >> 16;   // round-to-nearest-even
    return (unsigned short)u;
}
__device__ inline unsigned pack2_bf16(float a, float b) {
    return (unsigned)f2bf(a) | ((unsigned)f2bf(b) << 16);
}

// ---------------------------------------------------------------------------
// h = BN(x @ node_w + node_b)   (x:[N,3], node_w:[3,128])
// ---------------------------------------------------------------------------
__global__ __launch_bounds__(256)
void genconv_node_encode(const float* __restrict__ x,
                         const float* __restrict__ nw,
                         const float* __restrict__ nb,
                         const float* __restrict__ nbn,
                         float* __restrict__ resid) {
    int gid = blockIdx.x * 256 + threadIdx.x;
    int n = gid >> 7, c = gid & 127;
    float lin = x[n*3+0]*nw[c] + x[n*3+1]*nw[CCH+c] + x[n*3+2]*nw[2*CCH+c] + nb[c];
    float g = nbn[c], bb = nbn[CCH+c], m = nbn[2*CCH+c], v = nbn[3*CCH+c];
    resid[gid] = (lin - m) * g * rsqrtf(v + BN_EPS) + bb;
}

// ---------------------------------------------------------------------------
// Fold BN affine transforms into weights once:
//   ewF[j,c] = ew[j,c]*esc[c];  ebF[c] = (eb[c]-m)*esc[c] + b
//   s1[l,n]  = g*rsqrt(v+eps);  sh1[l,n] = (b1[l,n]-m)*s1 + b   (BN1 + bias fold)
// ---------------------------------------------------------------------------
__global__ __launch_bounds__(256)
void genconv_fold_params(const float* __restrict__ ew, const float* __restrict__ eb,
                         const float* __restrict__ ebn,
                         const float* __restrict__ b1, const float* __restrict__ bn1,
                         float* __restrict__ ewF, float* __restrict__ ebF,
                         float* __restrict__ s1,  float* __restrict__ sh1) {
    int id = blockIdx.x * 256 + threadIdx.x;
    if (id < 512) {
        int c = id & 127;
        float sc = ebn[c] * rsqrtf(ebn[3*CCH+c] + BN_EPS);
        ewF[id] = ew[id] * sc;
    } else if (id < 640) {
        int c = id - 512;
        float sc = ebn[c] * rsqrtf(ebn[3*CCH+c] + BN_EPS);
        ebF[c] = (eb[c] - ebn[2*CCH+c]) * sc + ebn[CCH+c];
    } else if (id < 640 + 768) {
        int k = id - 640;                 // l*256 + n
        int l = k >> 8, n = k & 255;
        const float* bp = bn1 + (size_t)l * 4 * HCH;
        float sc = bp[n] * rsqrtf(bp[3*HCH+n] + BN_EPS);
        s1[k]  = sc;
        sh1[k] = (b1[l*HCH+n] - bp[2*HCH+n]) * sc + bp[HCH+n];
    }
}

// ---------------------------------------------------------------------------
// Pre-swizzle W1/W2 (fp32) into per-lane WMMA bf16 B-fragment order.
// Fragment dword j of lane L in tile (kt,nt): K = kt*32 + (L/16)*16 + 2j{,+1},
// N = nt*16 + L%16.  Per tile: 32 lanes x 8 dwords contiguous.
// ---------------------------------------------------------------------------
__global__ __launch_bounds__(256)
void genconv_swizzle_w(const float* __restrict__ W1,
                       const float* __restrict__ W2,
                       unsigned* __restrict__ W1s,
                       unsigned* __restrict__ W2s) {
    int id   = blockIdx.x * 256 + threadIdx.x;   // 3 layers * 2 mats * 64 tiles * 256
    int j    = id & 7;
    int lane = (id >> 3) & 31;
    int t    = (id >> 8) & 63;
    int m    = (id >> 14) & 1;
    int l    = id >> 15;
    int half = lane >> 4, ln = lane & 15;
    if (m == 0) {                                   // W1: [3,128,256], kt<4, nt<16
        int kt = t >> 4, nt = t & 15;
        int K = kt*32 + half*16 + 2*j;
        int n = nt*16 + ln;
        const float* src = W1 + (l*CCH + K) * HCH + n;
        W1s[l*16384 + t*256 + lane*8 + j] = pack2_bf16(src[0], src[HCH]);
    } else {                                        // W2: [3,256,128], kt<8, nt<8
        int kt = t >> 3, nt = t & 7;
        int K = kt*32 + half*16 + 2*j;
        int n = nt*16 + ln;
        const float* src = W2 + (l*HCH + K) * CCH + n;
        W2s[l*16384 + t*256 + lane*8 + j] = pack2_bf16(src[0], src[CCH]);
    }
}

// ---------------------------------------------------------------------------
// zero mx/den/accum (laid out contiguously)
// ---------------------------------------------------------------------------
__global__ __launch_bounds__(256)
void genconv_clear(float* __restrict__ p) {
    p[blockIdx.x * 256 + threadIdx.x] = 0.0f;
}

// ---------------------------------------------------------------------------
// Two channels per edge thread; BN pre-folded into ewF/ebF.
// ---------------------------------------------------------------------------
__device__ inline void edge_msg2(int e, int c0,
                                 const float* __restrict__ resid,
                                 const int*   __restrict__ eidx,
                                 const float* __restrict__ eattr,
                                 const float* __restrict__ ewF,
                                 const float* __restrict__ ebF,
                                 int* dstOut, float* m0, float* m1) {
    int src = eidx[e];
    *dstOut = eidx[NEDGES + e];
    float ea0 = eattr[e*4+0], ea1 = eattr[e*4+1];
    float ea2 = eattr[e*4+2], ea3 = eattr[e*4+3];
    float2 lin = *(const float2*)&ebF[c0];
    float2 w0 = *(const float2*)&ewF[c0];
    float2 w1 = *(const float2*)&ewF[CCH   + c0];
    float2 w2 = *(const float2*)&ewF[2*CCH + c0];
    float2 w3 = *(const float2*)&ewF[3*CCH + c0];
    lin.x += ea0*w0.x + ea1*w1.x + ea2*w2.x + ea3*w3.x;
    lin.y += ea0*w0.y + ea1*w1.y + ea2*w2.y + ea3*w3.y;
    float2 rv = *(const float2*)&resid[src*CCH + c0];
    *m0 = fmaxf(rv.x + lin.x, 0.0f) + MSG_EPS;     // relu(x_j + e) + eps
    *m1 = fmaxf(rv.y + lin.y, 0.0f) + MSG_EPS;
}

__global__ __launch_bounds__(256)
void genconv_edge_max(const float* __restrict__ resid, const int* __restrict__ eidx,
                      const float* __restrict__ eattr,
                      const float* __restrict__ ewF, const float* __restrict__ ebF,
                      float* __restrict__ mx) {
    int gid = blockIdx.x * 256 + threadIdx.x;      // E * 64 threads
    int e = gid >> 6, c0 = (gid & 63) * 2;
    int dst; float m0, m1;
    edge_msg2(e, c0, resid, eidx, eattr, ewF, ebF, &dst, &m0, &m1);
    // s = msg*beta > 0 always -> uint order == float order
    atomicMax((unsigned*)&mx[dst*CCH + c0],     __float_as_uint(m0 * BETA));
    atomicMax((unsigned*)&mx[dst*CCH + c0 + 1], __float_as_uint(m1 * BETA));
}

__global__ __launch_bounds__(256)
void genconv_edge_sum(const float* __restrict__ resid, const int* __restrict__ eidx,
                      const float* __restrict__ eattr,
                      const float* __restrict__ ewF, const float* __restrict__ ebF,
                      const float* __restrict__ mx,
                      float* __restrict__ den, float* __restrict__ accum) {
    int gid = blockIdx.x * 256 + threadIdx.x;
    int e = gid >> 6, c0 = (gid & 63) * 2;
    int dst; float m0, m1;
    edge_msg2(e, c0, resid, eidx, eattr, ewF, ebF, &dst, &m0, &m1);
    float2 mv = *(const float2*)&mx[dst*CCH + c0];
    float ex0 = __expf(m0 * BETA - mv.x);
    float ex1 = __expf(m1 * BETA - mv.y);
    atomicAdd(&den[dst*CCH + c0],       ex0);
    atomicAdd(&den[dst*CCH + c0 + 1],   ex1);
    atomicAdd(&accum[dst*CCH + c0],     m0 * ex0);
    atomicAdd(&accum[dst*CCH + c0 + 1], m1 * ex1);
}

// ---------------------------------------------------------------------------
// Node update: c = resid + accum/den; hid = leaky(c@W1*s1 + sh1); resid += hid@W2+b2
// One wave per 16-row strip; 4 waves per block; v_wmma_f32_16x16x32_bf16.
// ---------------------------------------------------------------------------
__global__ __launch_bounds__(128)
void genconv_mlp_wmma(float* __restrict__ resid,
                      const float* __restrict__ accum,
                      const float* __restrict__ den,
                      const unsigned* __restrict__ W1s,
                      const unsigned* __restrict__ W2s,
                      const float* __restrict__ s1,
                      const float* __restrict__ sh1,
                      const float* __restrict__ b2,
                      int nNodes) {
    __shared__ unsigned cTile[4][16][64];     // 16 rows x 128 ch, bf16 pairs  (16 KB)
    __shared__ unsigned hidTile[4][16][128];  // 16 rows x 256 ch, bf16       (32 KB)

    const int tid  = threadIdx.x;
    const int w    = tid >> 5;
    const int lane = tid & 31;
    const int half = lane >> 4;
    const int ln   = lane & 15;
    const int rowBase = (blockIdx.x * 4 + w) * 16;

    // ---- stage C tile (resid + aggr) as bf16 ----
    for (int idx = lane; idx < 16 * 64; idx += 32) {
        int r = idx >> 6, dw = idx & 63;
        int gr = rowBase + r;
        unsigned packed = 0;
        if (gr < nNodes) {
            int base = gr * CCH + dw * 2;
            float2 d2 = *(const float2*)&den[base];
            float2 a2 = *(const float2*)&accum[base];
            float2 r2 = *(const float2*)&resid[base];
            float a0 = (d2.x > 0.0f) ? a2.x / d2.x : 0.0f;
            float a1 = (d2.y > 0.0f) ? a2.y / d2.y : 0.0f;
            packed = pack2_bf16(r2.x + a0, r2.y + a1);
        }
        cTile[w][r][dw] = packed;
    }
    __syncthreads();

    // ---- GEMM1: hid = leaky((c @ W1) * s1 + sh1) ----
    unsigned short* hs = (unsigned short*)&hidTile[w][0][0];
    for (int nt = 0; nt < 16; ++nt) {
        int n = nt * 16 + ln;
        v8f acc;
#pragma unroll
        for (int r = 0; r < 8; ++r) acc[r] = 0.0f;
#pragma unroll
        for (int kt = 0; kt < 4; ++kt) {
            BF16x16 A, B;
#pragma unroll
            for (int j = 0; j < 4; ++j) {   // A: K = kt*32 + half*16 + i
                A.u[j]     = cTile[w][ln][kt*16 + half*8 + j];
                A.u[4 + j] = cTile[w][ln][kt*16 + half*8 + 4 + j];
            }
            const uint4* bp = (const uint4*)(W1s + (kt*16 + nt)*256 + lane*8);
            __builtin_prefetch((const void*)(bp + 32), 0, 1);
            uint4 blo = bp[0], bhi = bp[1];
            B.u[0]=blo.x; B.u[1]=blo.y; B.u[2]=blo.z; B.u[3]=blo.w;
            B.u[4]=bhi.x; B.u[5]=bhi.y; B.u[6]=bhi.z; B.u[7]=bhi.w;
            acc = __builtin_amdgcn_wmma_f32_16x16x32_bf16(false, A.v, false, B.v,
                                                          (short)0, acc, false, false);
        }
        float scale = s1[n], shift = sh1[n];
#pragma unroll
        for (int r = 0; r < 8; ++r) {
            float hv = acc[r] * scale + shift;
            hv = (hv >= 0.0f) ? hv : 0.01f * hv;       // LeakyReLU(0.01)
            hs[(half*8 + r) * HCH + n] = f2bf(hv);     // D row M = half*8 + r
        }
    }
    __syncthreads();

    // ---- GEMM2: resid += hid @ W2 + b2 ----
    const unsigned* hp = (const unsigned*)&hidTile[w][0][0];
    for (int nt = 0; nt < 8; ++nt) {
        int n = nt * 16 + ln;
        float bias = b2[n];
        v8f acc;
#pragma unroll
        for (int r = 0; r < 8; ++r) acc[r] = bias;
#pragma unroll
        for (int kt = 0; kt < 8; ++kt) {
            BF16x16 A, B;
#pragma unroll
            for (int j = 0; j < 4; ++j) {
                A.u[j]     = hp[ln*128 + kt*16 + half*8 + j];
                A.u[4 + j] = hp[ln*128 + kt*16 + half*8 + 4 + j];
            }
            const uint4* bp = (const uint4*)(W2s + (kt*8 + nt)*256 + lane*8);
            __builtin_prefetch((const void*)(bp + 32), 0, 1);
            uint4 blo = bp[0], bhi = bp[1];
            B.u[0]=blo.x; B.u[1]=blo.y; B.u[2]=blo.z; B.u[3]=blo.w;
            B.u[4]=bhi.x; B.u[5]=bhi.y; B.u[6]=bhi.z; B.u[7]=bhi.w;
            acc = __builtin_amdgcn_wmma_f32_16x16x32_bf16(false, A.v, false, B.v,
                                                          (short)0, acc, false, false);
        }
#pragma unroll
        for (int r = 0; r < 8; ++r) {
            int gr = rowBase + half*8 + r;
            if (gr < nNodes) {
                int o = gr * CCH + n;
                resid[o] += acc[r];
            }
        }
    }
}

// ---------------------------------------------------------------------------
__global__ __launch_bounds__(256)
void genconv_sigmoid(const float* __restrict__ resid, float* __restrict__ out) {
    int gid = blockIdx.x * 256 + threadIdx.x;
    float x = resid[gid];
    out[gid] = 1.0f / (1.0f + __expf(-x));
}

// ---------------------------------------------------------------------------
extern "C" void kernel_launch(void* const* d_in, const int* in_sizes, int n_in,
                              void* d_out, int out_size, void* d_ws, size_t ws_size,
                              hipStream_t stream) {
    (void)in_sizes; (void)n_in; (void)out_size; (void)ws_size;
    const float* x     = (const float*)d_in[0];
    const int*   eidx  = (const int*)  d_in[1];
    const float* eattr = (const float*)d_in[2];
    const float* nw    = (const float*)d_in[3];
    const float* nb    = (const float*)d_in[4];
    const float* nbn   = (const float*)d_in[5];
    const float* ew    = (const float*)d_in[6];
    const float* eb    = (const float*)d_in[7];
    const float* ebn   = (const float*)d_in[8];
    const float* W1    = (const float*)d_in[9];
    const float* b1    = (const float*)d_in[10];
    const float* bn1   = (const float*)d_in[11];
    const float* W2    = (const float*)d_in[12];
    const float* b2    = (const float*)d_in[13];

    float* ws    = (float*)d_ws;
    float* resid = ws;                 // [N,128]
    float* mx    = ws + (size_t)NC;    // [N,128]  (mx,den,accum contiguous for clear)
    float* den   = ws + (size_t)2*NC;
    float* accum = ws + (size_t)3*NC;
    unsigned* W1s = (unsigned*)(ws + (size_t)4*NC);   // 3 * 16384 dwords
    unsigned* W2s = W1s + 3*16384;                    // 3 * 16384 dwords
    float* fp   = (float*)(W2s + 3*16384);
    float* ewF  = fp;                  // [4,128]
    float* ebF  = fp + 512;            // [128]
    float* s1   = fp + 640;            // [3,256]
    float* sh1  = fp + 640 + 768;      // [3,256]

    genconv_node_encode<<<NC/256, 256, 0, stream>>>(x, nw, nb, nbn, resid);
    genconv_fold_params<<<6,      256, 0, stream>>>(ew, eb, ebn, b1, bn1,
                                                    ewF, ebF, s1, sh1);
    genconv_swizzle_w  <<<384,    256, 0, stream>>>(W1, W2, W1s, W2s);

    const int edgeBlocks = (NEDGES * 64) / 256;    // 200000
    for (int l = 0; l < 3; ++l) {
        genconv_clear   <<<(3*NC)/256, 256, 0, stream>>>(mx);
        genconv_edge_max<<<edgeBlocks, 256, 0, stream>>>(resid, eidx, eattr, ewF, ebF, mx);
        genconv_edge_sum<<<edgeBlocks, 256, 0, stream>>>(resid, eidx, eattr, ewF, ebF,
                                                         mx, den, accum);
        genconv_mlp_wmma<<<(NNODES + 63)/64, 128, 0, stream>>>(
            resid, accum, den,
            W1s + l*16384, W2s + l*16384,
            s1 + l*HCH, sh1 + l*HCH, b2 + l*CCH, NNODES);
    }
    genconv_sigmoid<<<NC/256, 256, 0, stream>>>(resid, (float*)d_out);
}
